// AdaptiveDepthRWKV_38036230373515
// MI455X (gfx1250) — matrix operand order repaired and checked by reading, not compiled, not run on hardware
//
#include <hip/hip_runtime.h>
#include <hip/hip_bf16.h>
#include <math.h>

// Model dims (fixed by the reference)
#define B_N   2
#define T_N   1024
#define D_N   1024
#define DFF_N 4096
#define V_N   32000
#define L_N   12
#define BT_N  (B_N * T_N)

// GEMM tiling
#define GEMM_BM 256   // block M tile (8 waves x 32 rows)
#define GEMM_BN 64    // block N tile (4 x 16)

typedef __attribute__((ext_vector_type(16))) _Float16 v16h;
typedef __attribute__((ext_vector_type(8)))  _Float16 v8h;
typedef __attribute__((ext_vector_type(4)))  _Float16 v4h;
typedef __attribute__((ext_vector_type(8)))  float    v8f;

__device__ __forceinline__ float sigmoid_f(float x) { return 1.0f / (1.0f + __expf(-x)); }

// ---------------------------------------------------------------------------
// CDNA5 async global->LDS copy (ASYNCcnt-tracked), CDNA5 ISA
// Each lane copies 16 bytes: LDS[lds_off] = MEM[gptr].
// ---------------------------------------------------------------------------
__device__ __forceinline__ void async_load_lds_b128(unsigned lds_off, const void* gptr) {
  asm volatile("global_load_async_to_lds_b128 %0, %1, off"
               :: "v"(lds_off), "v"(gptr) : "memory");
}
__device__ __forceinline__ void wait_async_le1() {
  asm volatile("s_wait_asynccnt 0x1" ::: "memory");
}
__device__ __forceinline__ void wait_async_le0() {
  asm volatile("s_wait_asynccnt 0x0" ::: "memory");
}

// ---------------------------------------------------------------------------
// fp32 -> f16 bulk conversion (vectorized 4-wide)
// ---------------------------------------------------------------------------
__global__ __launch_bounds__(256) void cvt_f32_f16_kernel(
    const float4* __restrict__ in, v4h* __restrict__ out, size_t n4)
{
  for (size_t i = (size_t)blockIdx.x * 256 + threadIdx.x; i < n4;
       i += (size_t)gridDim.x * 256) {
    float4 v = in[i];
    v4h o;
    o[0] = (_Float16)v.x; o[1] = (_Float16)v.y;
    o[2] = (_Float16)v.z; o[3] = (_Float16)v.w;
    out[i] = o;
  }
}

// ---------------------------------------------------------------------------
// WMMA GEMM (f16 inputs, f32 accumulate): OUT[M,N] = A[M,K] @ W[N,K]^T
//   mode 0: OUTF = acc
//   mode 1: OUTF = acc + AUX[m,n]              (residual; AUX may alias OUTF)
//   mode 2: OUTH = f16( silu(AUX[m,n]) * acc ) (SwiGLU gate)
// Block: 256 thr = 8 waves; block tile 256(M) x 64(N); wave tile 32 x 64.
// B tile (64x32 f16, 4KB) staged to LDS with async copies, double buffered;
// shared by all 8 waves. A fragments loaded direct from global (L2-resident).
// A/B use contiguous k order (k = 16*grp + s); C/D layout per CDNA5 ISA §7.12.2 (wave32).
// ---------------------------------------------------------------------------
__global__ __launch_bounds__(256) void gemm_f16_wmma(
    const _Float16* __restrict__ A, const _Float16* __restrict__ W,
    float* __restrict__ OUTF, _Float16* __restrict__ OUTH,
    const float* __restrict__ AUX,
    int M, int N, int K, int mode)
{
  __shared__ __align__(32) _Float16 bsh[2][GEMM_BN * 32];  // 2 x 4KB

  const int tid  = threadIdx.x;
  const int lane = tid & 31;
  const int wave = tid >> 5;
  const int r16  = lane & 15;
  const int grp  = lane >> 4;

  const int m0 = blockIdx.x * GEMM_BM + wave * 32;
  const int n0 = blockIdx.y * GEMM_BN;

  // Staging assignment: thread -> (row, 8-half segment): 64 rows x 4 segs
  const int srow = tid >> 2;
  const int sseg = (tid & 3) * 8;
  const _Float16* wsrc = W + (size_t)(n0 + srow) * K + sseg;
  const unsigned lds0 = (unsigned)(size_t)(&bsh[0][srow * 32 + sseg]);
  const unsigned lds1 = (unsigned)(size_t)(&bsh[1][srow * 32 + sseg]);

  const _Float16* arow0 = A + (size_t)(m0 + r16) * K;
  const _Float16* arow1 = A + (size_t)(m0 + 16 + r16) * K;

  v8f acc[2][4] = {};

  // Prologue: stage k0 = 0 into buffer 0
  async_load_lds_b128(lds0, wsrc);

  for (int k0 = 0, it = 0; k0 < K; k0 += 32, ++it) {
    const int  cur     = it & 1;
    const bool hasNext = (k0 + 32) < K;
    if (hasNext) {
      async_load_lds_b128(cur ? lds0 : lds1, wsrc + k0 + 32);
      wait_async_le1();            // current buffer's copy complete (this wave)
    } else {
      wait_async_le0();
    }
    __syncthreads();               // all waves' copies complete

    // A fragments: halves [0..15] <- K=k0+grp*16..k0+grp*16+15
    v8h a0lo = *(const v8h*)(arow0 + k0 + grp * 16);
    v8h a0hi = *(const v8h*)(arow0 + k0 + grp * 16 + 8);
    v8h a1lo = *(const v8h*)(arow1 + k0 + grp * 16);
    v8h a1hi = *(const v8h*)(arow1 + k0 + grp * 16 + 8);
    v16h a0 = __builtin_shufflevector(a0lo, a0hi, 0,1,2,3,4,5,6,7,8,9,10,11,12,13,14,15);
    v16h a1 = __builtin_shufflevector(a1lo, a1hi, 0,1,2,3,4,5,6,7,8,9,10,11,12,13,14,15);

    const _Float16* bs = &bsh[cur][0];
#pragma unroll
    for (int j = 0; j < 4; ++j) {
      // B (32x16): lane col = r16; grp0 -> K 0..15, grp1 -> K 16..31
      v16h b = *(const v16h*)(bs + (j * 16 + r16) * 32 + grp * 16);
      acc[0][j] = __builtin_amdgcn_wmma_f32_16x16x32_f16(
          false, a0, false, b, (short)0, acc[0][j], false, false);
      acc[1][j] = __builtin_amdgcn_wmma_f32_16x16x32_f16(
          false, a1, false, b, (short)0, acc[1][j], false, false);
    }
    __syncthreads();               // protect buffer being re-staged next iter
  }

  // C/D layout: VGPR r -> row (grp*8 + r); col = n0 + j*16 + r16
#pragma unroll
  for (int h = 0; h < 2; ++h) {
    const int mbase = m0 + h * 16 + grp * 8;
#pragma unroll
    for (int j = 0; j < 4; ++j) {
      const int col = n0 + j * 16 + r16;
#pragma unroll
      for (int r = 0; r < 8; ++r) {
        const size_t o = (size_t)(mbase + r) * N + col;
        float v = acc[h][j][r];
        if (mode == 1) {
          OUTF[o] = v + AUX[o];
        } else if (mode == 2) {
          float g = AUX[o];
          OUTH[o] = (_Float16)((g * sigmoid_f(g)) * v);
        } else {
          OUTF[o] = v;
        }
      }
    }
  }
}

// ---------------------------------------------------------------------------
// Row LayerNorm; gather != nullptr => embedding lookup; outh: optional f16 copy
// ---------------------------------------------------------------------------
__global__ __launch_bounds__(256) void ln_rows_kernel(
    const float* __restrict__ in, const int* __restrict__ gather,
    float* __restrict__ out, _Float16* __restrict__ outh,
    const float* __restrict__ w, const float* __restrict__ bb, int D)
{
  const int row = blockIdx.x;
  const float* p = gather ? (in + (size_t)gather[row] * D)
                          : (in + (size_t)row * D);
  float s = 0.f, ss = 0.f;
  for (int i = threadIdx.x; i < D; i += 256) { float v = p[i]; s += v; ss += v * v; }
  __shared__ float rs[256], rq[256];
  rs[threadIdx.x] = s; rq[threadIdx.x] = ss; __syncthreads();
  for (int o = 128; o > 0; o >>= 1) {
    if (threadIdx.x < o) { rs[threadIdx.x] += rs[threadIdx.x + o];
                           rq[threadIdx.x] += rq[threadIdx.x + o]; }
    __syncthreads();
  }
  const float mean = rs[0] / (float)D;
  const float var  = rq[0] / (float)D - mean * mean;
  const float inv  = rsqrtf(var + 1e-5f);
  float*    q  = out + (size_t)row * D;
  _Float16* qh = outh ? (outh + (size_t)row * D) : nullptr;
  for (int i = threadIdx.x; i < D; i += 256) {
    float v = (p[i] - mean) * inv * w[i] + bb[i];
    q[i] = v;
    if (qh) qh[i] = (_Float16)v;
  }
}

// ---------------------------------------------------------------------------
// RWKV time-mix recurrence: out[t] = sigmoid(r[t]) * (sum_{s<=t} kv[s]*dec^{t-s})
// ---------------------------------------------------------------------------
__global__ void rwkv_scan_kernel(
    const float* __restrict__ r, const float* __restrict__ k,
    const float* __restrict__ v, const float* __restrict__ decay,
    float* __restrict__ out, int T, int D)
{
  const int tid = blockIdx.x * blockDim.x + threadIdx.x;  // [0, B*D)
  const int b = tid / D, d = tid % D;
  const float dec = sigmoid_f(decay[d]);
  float s = 0.f;
  size_t idx = (size_t)b * T * D + d;
  for (int t = 0; t < T; ++t, idx += D) {
    s = s * dec + k[idx] * v[idx];
    out[idx] = sigmoid_f(r[idx]) * s;
  }
}

// ---------------------------------------------------------------------------
// GroupNorm(1, D) over (T,D) per batch: deterministic two-stage reduction.
// ---------------------------------------------------------------------------
__global__ __launch_bounds__(256) void gn_partial_kernel(
    const float* __restrict__ x, float* __restrict__ part,
    int perBatch, int blocksPerBatch)
{
  const int b = blockIdx.x / blocksPerBatch;
  const int c = blockIdx.x % blocksPerBatch;
  const int chunk = perBatch / blocksPerBatch;
  const float* p = x + (size_t)b * perBatch + (size_t)c * chunk;
  float s = 0.f, ss = 0.f;
  for (int i = threadIdx.x; i < chunk; i += 256) { float v = p[i]; s += v; ss += v * v; }
  __shared__ float rs[256], rq[256];
  rs[threadIdx.x] = s; rq[threadIdx.x] = ss; __syncthreads();
  for (int o = 128; o > 0; o >>= 1) {
    if (threadIdx.x < o) { rs[threadIdx.x] += rs[threadIdx.x + o];
                           rq[threadIdx.x] += rq[threadIdx.x + o]; }
    __syncthreads();
  }
  if (threadIdx.x == 0) { part[2 * blockIdx.x] = rs[0]; part[2 * blockIdx.x + 1] = rq[0]; }
}

__global__ __launch_bounds__(256) void gn_finalize_kernel(
    const float* __restrict__ part, float* __restrict__ stats,
    int blocksPerBatch, int perBatch)
{
  __shared__ float rs[256], rq[256];
  for (int b = 0; b < B_N; ++b) {
    float s = 0.f, ss = 0.f;
    for (int i = threadIdx.x; i < blocksPerBatch; i += 256) {
      s  += part[2 * (b * blocksPerBatch + i)];
      ss += part[2 * (b * blocksPerBatch + i) + 1];
    }
    rs[threadIdx.x] = s; rq[threadIdx.x] = ss; __syncthreads();
    for (int o = 128; o > 0; o >>= 1) {
      if (threadIdx.x < o) { rs[threadIdx.x] += rs[threadIdx.x + o];
                             rq[threadIdx.x] += rq[threadIdx.x + o]; }
      __syncthreads();
    }
    if (threadIdx.x == 0) {
      float mean = rs[0] / (float)perBatch;
      float var  = rq[0] / (float)perBatch - mean * mean;
      stats[2 * b] = mean;
      stats[2 * b + 1] = rsqrtf(var + 1e-5f);
    }
    __syncthreads();
  }
}

// Normalize + affine, emit f16 (consumed only as a GEMM A operand)
__global__ void gn_apply_kernel(
    const float* __restrict__ x, _Float16* __restrict__ outh,
    const float* __restrict__ stats, const float* __restrict__ w,
    const float* __restrict__ bb, int total, int perBatch, int D)
{
  for (size_t i = (size_t)blockIdx.x * blockDim.x + threadIdx.x; i < (size_t)total;
       i += (size_t)gridDim.x * blockDim.x) {
    int b = (int)(i / perBatch);
    int d = (int)(i % D);
    outh[i] = (_Float16)((x[i] - stats[2 * b]) * stats[2 * b + 1] * w[d] + bb[d]);
  }
}

// ---------------------------------------------------------------------------
// Per-token softmax stats over V: CE to target, argmax, entropy.
// ---------------------------------------------------------------------------
__global__ __launch_bounds__(256) void token_stats_kernel(
    const float* __restrict__ logits, const int* __restrict__ targets,
    float* __restrict__ ce, int* __restrict__ pred, float* __restrict__ ent, int V)
{
  const int tkn = blockIdx.x;
  const float* row = logits + (size_t)tkn * V;
  __shared__ float sa[256], sb[256];
  __shared__ int   si[256];

  float mx = -3.4e38f; int ai = 0;
  for (int i = threadIdx.x; i < V; i += 256) {
    float l = row[i];
    if (l > mx) { mx = l; ai = i; }
  }
  sa[threadIdx.x] = mx; si[threadIdx.x] = ai; __syncthreads();
  for (int o = 128; o > 0; o >>= 1) {
    if (threadIdx.x < o) {
      float m2 = sa[threadIdx.x + o]; int i2 = si[threadIdx.x + o];
      if (m2 > sa[threadIdx.x] || (m2 == sa[threadIdx.x] && i2 < si[threadIdx.x])) {
        sa[threadIdx.x] = m2; si[threadIdx.x] = i2;
      }
    }
    __syncthreads();
  }
  const float gmax = sa[0];
  const int   gidx = si[0];
  __syncthreads();

  float s1 = 0.f, s2 = 0.f;
  for (int i = threadIdx.x; i < V; i += 256) {
    float l = row[i];
    float e = __expf(l - gmax);
    s1 += e; s2 += e * l;
  }
  sa[threadIdx.x] = s1; sb[threadIdx.x] = s2; __syncthreads();
  for (int o = 128; o > 0; o >>= 1) {
    if (threadIdx.x < o) { sa[threadIdx.x] += sa[threadIdx.x + o];
                           sb[threadIdx.x] += sb[threadIdx.x + o]; }
    __syncthreads();
  }
  if (threadIdx.x == 0) {
    float S1 = sa[0], S2 = sb[0];
    float lse = gmax + logf(S1);
    ce[tkn]   = -(row[targets[tkn]] - lse);
    pred[tkn] = gidx;
    ent[tkn]  = lse - S2 / S1;
  }
}

// he.mean over T per (b,d)
__global__ void he_mean_kernel(const float* __restrict__ h, float* __restrict__ hm,
                               int T, int D)
{
  const int tid = blockIdx.x * blockDim.x + threadIdx.x;  // [0, B*D)
  const int b = tid / D, d = tid % D;
  float s = 0.f;
  size_t idx = (size_t)b * T * D + d;
  for (int t = 0; t < T; ++t, idx += D) s += h[idx];
  hm[tid] = s / (float)T;
}

// Gate MLP: g = relu(hm @ w1^T + b1) [B,64]; conf = sigmoid(g @ w2^T + b2) [B]
__global__ __launch_bounds__(128) void gate_kernel(
    const float* __restrict__ hm, const float* __restrict__ w1,
    const float* __restrict__ b1, const float* __restrict__ w2,
    const float* __restrict__ b2, float* __restrict__ conf, int D)
{
  __shared__ float g[128];
  const int u = threadIdx.x % 64;
  const int b = threadIdx.x / 64;
  float s = b1[u];
  const float* hb = hm + (size_t)b * D;
  const float* wr = w1 + (size_t)u * D;
  for (int d = 0; d < D; ++d) s += hb[d] * wr[d];
  g[threadIdx.x] = fmaxf(s, 0.f);
  __syncthreads();
  if (threadIdx.x < B_N) {
    const int bb = threadIdx.x;
    float t = b2[0];
    for (int uu = 0; uu < 64; ++uu) t += g[bb * 64 + uu] * w2[uu];
    conf[bb] = sigmoid_f(t);
  }
}

// ---------------------------------------------------------------------------
// Final loss assembly (single block).
// ---------------------------------------------------------------------------
__device__ float block_sum256(float v, float* buf) {
  buf[threadIdx.x] = v; __syncthreads();
  for (int o = 128; o > 0; o >>= 1) {
    if (threadIdx.x < o) buf[threadIdx.x] += buf[threadIdx.x + o];
    __syncthreads();
  }
  float r = buf[0]; __syncthreads();
  return r;
}

__device__ float exit_contrib(const float* ce, const int* pe, const float* ent,
                              const float* conf, const int* pf, float wj,
                              int T, int BT, float inv_max_ent, float* buf)
{
  float sc = 0.f, a0 = 0.f, a1 = 0.f, so = 0.f;
  for (int i = threadIdx.x; i < BT; i += 256) {
    sc += ce[i];
    int agree = (pe[i] == pf[i]) ? 1 : 0;
    if (i < T) a0 += (float)agree; else a1 += (float)agree;
    so += (1.f - ent[i] * inv_max_ent) * (agree ? 0.f : 1.f);
  }
  sc = block_sum256(sc, buf);
  a0 = block_sum256(a0, buf);
  a1 = block_sum256(a1, buf);
  so = block_sum256(so, buf);
  float loss = wj * sc / (float)BT + 0.1f * so / (float)BT;
  float ag[2] = { a0 / (float)T, a1 / (float)T };
  float bce = 0.f;
  for (int b = 0; b < B_N; ++b) {
    float c = fminf(fmaxf(conf[b], 1e-7f), 1.f - 1e-7f);
    bce += -(ag[b] * logf(c) + (1.f - ag[b]) * logf(1.f - c));
  }
  loss += 0.5f * (bce / (float)B_N);
  return loss;
}

__global__ __launch_bounds__(256) void loss_kernel(
    const float* __restrict__ ce_f, const int* __restrict__ pred_f,
    const float* __restrict__ ce0, const int* __restrict__ pred0,
    const float* __restrict__ ent0, const float* __restrict__ conf0,
    const float* __restrict__ ce1, const int* __restrict__ pred1,
    const float* __restrict__ ent1, const float* __restrict__ conf1,
    float* __restrict__ out_loss, int T, int BT)
{
  __shared__ float buf[256];
  const float inv_max_ent = 1.f / logf((float)V_N);
  float sc = 0.f;
  for (int i = threadIdx.x; i < BT; i += 256) sc += ce_f[i];
  sc = block_sum256(sc, buf);
  float loss = 1.0f * sc / (float)BT;  // WEIGHTS[-1] = 1.0
  loss += exit_contrib(ce0, pred0, ent0, conf0, pred_f, 0.3f, T, BT, inv_max_ent, buf);
  loss += exit_contrib(ce1, pred1, ent1, conf1, pred_f, 0.5f, T, BT, inv_max_ent, buf);
  if (threadIdx.x == 0) out_loss[0] = loss;
}

// ---------------------------------------------------------------------------
// Host-side orchestration
// ---------------------------------------------------------------------------
static inline void cvt_launch(const float* src, _Float16* dst, size_t n, hipStream_t s) {
  cvt_f32_f16_kernel<<<2048, 256, 0, s>>>((const float4*)src, (v4h*)dst, n / 4);
}

extern "C" void kernel_launch(void* const* d_in, const int* in_sizes, int n_in,
                              void* d_out, int out_size, void* d_ws, size_t ws_size,
                              hipStream_t stream)
{
  const int    B = B_N, T = T_N, D = D_N, DFF = DFF_N, V = V_N, BT = BT_N;
  (void)in_sizes; (void)n_in; (void)out_size; (void)ws_size;

  const int*   idx      = (const int*)  d_in[0];
  const int*   targets  = (const int*)  d_in[1];
  const float* embed    = (const float*)d_in[2];
  const float* ln_in_w  = (const float*)d_in[3];
  const float* ln_in_b  = (const float*)d_in[4];
  const float* Wr       = (const float*)d_in[5];
  const float* Wk       = (const float*)d_in[6];
  const float* Wv       = (const float*)d_in[7];
  const float* Wo_tm    = (const float*)d_in[8];
  const float* decay    = (const float*)d_in[9];
  const float* gn_w     = (const float*)d_in[10];
  const float* gn_b     = (const float*)d_in[11];
  const float* ln1_w    = (const float*)d_in[12];
  const float* ln1_b    = (const float*)d_in[13];
  const float* ln2_w    = (const float*)d_in[14];
  const float* ln2_b    = (const float*)d_in[15];
  const float* W1       = (const float*)d_in[16];
  const float* W2       = (const float*)d_in[17];
  const float* Wo_cm    = (const float*)d_in[18];
  const float* ln_out_w = (const float*)d_in[19];
  const float* ln_out_b = (const float*)d_in[20];
  const float* exit_ln_w= (const float*)d_in[21];
  const float* exit_ln_b= (const float*)d_in[22];
  const float* exit_head= (const float*)d_in[23];
  const float* gate_w1  = (const float*)d_in[24];
  const float* gate_b1  = (const float*)d_in[25];
  const float* gate_w2  = (const float*)d_in[26];
  const float* gate_b2  = (const float*)d_in[27];

  float* out_logits = (float*)d_out;                // [BT,V]: exit scratch, then final
  float* out_loss   = out_logits + (size_t)BT * V;  // [1]

  // Workspace carving (~175 MB)
  float* p = (float*)d_ws;
  float* x    = p; p += (size_t)BT * D;
  float* h    = p; p += (size_t)BT * D;
  float* rb   = p; p += (size_t)BT * D;
  float* kb   = p; p += (size_t)BT * D;
  float* vb   = p; p += (size_t)BT * D;
  float* tm   = p; p += (size_t)BT * D;
  float* ab   = p; p += (size_t)BT * DFF;
  _Float16* hh  = (_Float16*)p; p += (size_t)BT * D / 2;     // f16 LN output
  _Float16* tmh = (_Float16*)p; p += (size_t)BT * D / 2;     // f16 GN output
  _Float16* abh = (_Float16*)p; p += (size_t)BT * DFF / 2;   // f16 SwiGLU gate
  _Float16* wh  = (_Float16*)p; p += (size_t)V * D / 2;      // f16 weight staging (max V*D)
  float* hm   = p; p += (size_t)B * D;
  float* part = p; p += (size_t)B * 128 * 2;
  float* stats= p; p += 4;
  float* cef  = p; p += BT;
  float* entf = p; p += BT;
  float* ce0  = p; p += BT;
  float* ent0 = p; p += BT;
  float* ce1  = p; p += BT;
  float* ent1 = p; p += BT;
  float* conf = p; p += 4;
  int* predf  = (int*)p;
  int* pred0  = predf + BT;
  int* pred1  = pred0 + BT;

  const dim3 blk(256);
  const dim3 gD (BT / GEMM_BM, D   / GEMM_BN);   // N = D
  const dim3 gF (BT / GEMM_BM, DFF / GEMM_BN);   // N = DFF
  const dim3 gV (BT / GEMM_BM, V   / GEMM_BN);   // N = V

  // x = LN(embed[idx])
  ln_rows_kernel<<<BT, blk, 0, stream>>>(embed, idx, x, nullptr, ln_in_w, ln_in_b, D);

  for (int l = 0; l < L_N; ++l) {
    const float* Wr_l  = Wr    + (size_t)l * D * D;
    const float* Wk_l  = Wk    + (size_t)l * D * D;
    const float* Wv_l  = Wv    + (size_t)l * D * D;
    const float* Wo_l  = Wo_tm + (size_t)l * D * D;
    const float* W1_l  = W1    + (size_t)l * DFF * D;
    const float* W2_l  = W2    + (size_t)l * DFF * D;
    const float* Wcm_l = Wo_cm + (size_t)l * D * DFF;

    // --- Time-mix ---
    ln_rows_kernel<<<BT, blk, 0, stream>>>(x, nullptr, h, hh, ln1_w + l * D, ln1_b + l * D, D);
    cvt_launch(Wr_l, wh, (size_t)D * D, stream);
    gemm_f16_wmma<<<gD, blk, 0, stream>>>(hh, wh, rb, nullptr, nullptr, BT, D, D, 0);
    cvt_launch(Wk_l, wh, (size_t)D * D, stream);
    gemm_f16_wmma<<<gD, blk, 0, stream>>>(hh, wh, kb, nullptr, nullptr, BT, D, D, 0);
    cvt_launch(Wv_l, wh, (size_t)D * D, stream);
    gemm_f16_wmma<<<gD, blk, 0, stream>>>(hh, wh, vb, nullptr, nullptr, BT, D, D, 0);
    rwkv_scan_kernel<<<(B * D) / 256, blk, 0, stream>>>(rb, kb, vb, decay + l * D, tm, T, D);
    gn_partial_kernel<<<B * 128, blk, 0, stream>>>(tm, part, T * D, 128);
    gn_finalize_kernel<<<1, blk, 0, stream>>>(part, stats, 128, T * D);
    gn_apply_kernel<<<2048, blk, 0, stream>>>(tm, tmh, stats, gn_w + l * D, gn_b + l * D,
                                              B * T * D, T * D, D);
    cvt_launch(Wo_l, wh, (size_t)D * D, stream);
    gemm_f16_wmma<<<gD, blk, 0, stream>>>(tmh, wh, x, nullptr, x, BT, D, D, 1);

    // --- Channel-mix (SwiGLU) ---
    ln_rows_kernel<<<BT, blk, 0, stream>>>(x, nullptr, h, hh, ln2_w + l * D, ln2_b + l * D, D);
    cvt_launch(W1_l, wh, (size_t)DFF * D, stream);
    gemm_f16_wmma<<<gF, blk, 0, stream>>>(hh, wh, ab, nullptr, nullptr, BT, DFF, D, 0);
    cvt_launch(W2_l, wh, (size_t)DFF * D, stream);
    gemm_f16_wmma<<<gF, blk, 0, stream>>>(hh, wh, nullptr, abh, ab, BT, DFF, D, 2);
    cvt_launch(Wcm_l, wh, (size_t)D * DFF, stream);
    gemm_f16_wmma<<<gD, blk, 0, stream>>>(abh, wh, x, nullptr, x, BT, D, DFF, 1);

    // --- Early exit heads (layers 4, 8) ---
    const int e = (l == 3) ? 0 : (l == 7) ? 1 : -1;
    if (e >= 0) {
      ln_rows_kernel<<<BT, blk, 0, stream>>>(x, nullptr, h, hh,
                                             exit_ln_w + e * D, exit_ln_b + e * D, D);
      cvt_launch(exit_head + (size_t)e * V * D, wh, (size_t)V * D, stream);
      gemm_f16_wmma<<<gV, blk, 0, stream>>>(hh, wh, out_logits, nullptr, nullptr, BT, V, D, 0);
      token_stats_kernel<<<BT, blk, 0, stream>>>(out_logits, targets,
                                                 (e == 0) ? ce0 : ce1,
                                                 (e == 0) ? pred0 : pred1,
                                                 (e == 0) ? ent0 : ent1, V);
      he_mean_kernel<<<(B * D) / 256, blk, 0, stream>>>(h, hm, T, D);
      gate_kernel<<<1, 128, 0, stream>>>(hm, gate_w1 + (size_t)e * 64 * D,
                                         gate_b1 + e * 64, gate_w2 + e * 64,
                                         gate_b2 + e, conf + e * 2, D);
    }
  }

  // Final head (tied embedding) overwrites the logits region.
  ln_rows_kernel<<<BT, blk, 0, stream>>>(x, nullptr, h, hh, ln_out_w, ln_out_b, D);
  cvt_launch(embed, wh, (size_t)V * D, stream);
  gemm_f16_wmma<<<gV, blk, 0, stream>>>(hh, wh, out_logits, nullptr, nullptr, BT, V, D, 0);
  token_stats_kernel<<<BT, blk, 0, stream>>>(out_logits, targets, cef, predf, entf, V);

  loss_kernel<<<1, blk, 0, stream>>>(cef, predf, ce0, pred0, ent0, conf + 0,
                                     ce1, pred1, ent1, conf + 2, out_loss, T, BT);
}
